// BranchedCrossAttnProcessor_NS2_62809601737283
// MI455X (gfx1250) — matrix operand order repaired and checked by reading, not compiled, not run on hardware
//
#include <hip/hip_runtime.h>

typedef __attribute__((ext_vector_type(16))) _Float16 v16h;
typedef __attribute__((ext_vector_type(8)))  _Float16 v8h;
typedef __attribute__((ext_vector_type(8)))  float    v8f;

#define S_LEN   4096
#define HDIM    1280
#define CDIM    2048
#define TLEN    77
#define ILEN    32
#define NB      4
#define NHEADS  20
#define DH      64
#define LP      128
#define LVALID  109

// Assemble a 16x32 f16 A-fragment from two contiguous 8-half chunks.
// Contiguous K order: lane-half holds K=kbase+0..7 (VGPR0-3) and
// K=kbase+8..15 (VGPR4-7), kbase = 0 (lanes 0-15) or 16 (lanes 16-31).
__device__ __forceinline__ v16h make_a(const v8h lo, const v8h hi) {
  v16h a;
#pragma unroll
  for (int i = 0; i < 8; ++i) { a[i] = lo[i]; a[i + 8] = hi[i]; }
  return a;
}

// ---------------------------------------------------------------------------
// Packing / layout kernels
// ---------------------------------------------------------------------------
__global__ void f32_to_f16_kernel(const float* __restrict__ src,
                                  _Float16* __restrict__ dst, long n) {
  long i = (long)blockIdx.x * blockDim.x + threadIdx.x;
  if (i < n) dst[i] = (_Float16)src[i];
}

// W [R][C] (row-major) -> WT [C][R] f16  (so B^T rows are contiguous in K)
__global__ void transT_f16_kernel(const float* __restrict__ W,
                                  _Float16* __restrict__ WT, int R, int C) {
  long i = (long)blockIdx.x * blockDim.x + threadIdx.x;
  if (i < (long)R * C) {
    int r = (int)(i / C);
    int c = (int)(i % C);
    WT[(long)c * R + r] = (_Float16)W[i];
  }
}

// Concatenated transposed weights: WT [1280][4096], k<2048 -> W0, else W1.
__global__ void catT_f16_kernel(const float* __restrict__ W0,
                                const float* __restrict__ W1,
                                _Float16* __restrict__ WT) {
  long i = (long)blockIdx.x * blockDim.x + threadIdx.x;
  if (i < (long)HDIM * 4096) {
    int n = (int)(i >> 12);       // / 4096
    int k = (int)(i & 4095);
    float v = (k < CDIM) ? W0[(long)k * HDIM + n]
                         : W1[(long)(k - CDIM) * HDIM + n];
    WT[i] = (_Float16)v;
  }
}

// kvA [4][128][4096] f16: rows 0..76 = ehs (cols 0..2047), rows 77..108 =
// id_embedding[b%2] (cols 2048..4095), everything else zero.
__global__ void kva_pack_kernel(const float* __restrict__ ehs,
                                const float* __restrict__ idemb,
                                _Float16* __restrict__ kvA) {
  long i = (long)blockIdx.x * blockDim.x + threadIdx.x;
  if (i < (long)NB * LP * 4096) {
    int k    = (int)(i & 4095);
    int row  = (int)((i >> 12) & 127);
    int bidx = (int)(i >> 19);
    float v = 0.0f;
    if (row < TLEN && k < CDIM)
      v = ehs[((long)bidx * TLEN + row) * CDIM + k];
    else if (row >= TLEN && row < LVALID && k >= CDIM)
      v = idemb[((long)(bidx & 1) * ILEN + (row - TLEN)) * CDIM + (k - CDIM)];
    kvA[i] = (_Float16)v;
  }
}

// ---------------------------------------------------------------------------
// Generic WMMA GEMM: C[M,N] = A[M,K] * B[K,N], with BT = B^T [N][K] f16.
// One wave per block, 32(M) x 128(N) tile, K-step 32: 2 A-fragments,
// 16 accumulators, 16 WMMAs per K-step (B-fragment reused in registers).
// OUTMODE: 0 = f32 row-major + bias, 1 = f16 row-major, 2 = f16 transposed.
// ---------------------------------------------------------------------------
template <int OUTMODE>
__global__ void __launch_bounds__(32)
gemm_wmma(const _Float16* __restrict__ A, const _Float16* __restrict__ BT,
          void* __restrict__ Cout, const float* __restrict__ bias,
          int M, int N, int K) {
  const int lane  = threadIdx.x;
  const int lm    = lane & 15;
  const int half  = lane >> 4;
  const int kbase = half << 4;   // 0 or 16 (A fragment)
  const int koff  = half << 4;   // 0 or 16 (B fragment)
  const int n0    = blockIdx.y * 128;
  const int m0    = blockIdx.x * 32;

  v8f acc[2][8];
#pragma unroll
  for (int t = 0; t < 2; ++t)
#pragma unroll
    for (int j = 0; j < 8; ++j)
#pragma unroll
      for (int i = 0; i < 8; ++i) acc[t][j][i] = 0.0f;

  const _Float16* arow0 = A + (long)(m0 + lm) * K;
  const _Float16* arow1 = A + (long)(m0 + 16 + lm) * K;
  for (int kk = 0; kk < K; kk += 32) {
    const v16h a0 = make_a(*(const v8h*)(arow0 + kk + kbase),
                           *(const v8h*)(arow0 + kk + kbase + 8));
    const v16h a1 = make_a(*(const v8h*)(arow1 + kk + kbase),
                           *(const v8h*)(arow1 + kk + kbase + 8));
    __builtin_prefetch(arow0 + kk + 64, 0, 3);
    __builtin_prefetch(arow1 + kk + 64, 0, 3);
#pragma unroll
    for (int j = 0; j < 8; ++j) {
      const _Float16* brow = BT + (long)(n0 + j * 16 + lm) * K + kk + koff;
      const v16h b = *(const v16h*)brow;
      acc[0][j] = __builtin_amdgcn_wmma_f32_16x16x32_f16(
          false, a0, false, b, (short)0, acc[0][j], false, false);
      acc[1][j] = __builtin_amdgcn_wmma_f32_16x16x32_f16(
          false, a1, false, b, (short)0, acc[1][j], false, false);
    }
  }

#pragma unroll
  for (int t = 0; t < 2; ++t) {
    const int rbase = m0 + t * 16 + (half << 3);
#pragma unroll
    for (int j = 0; j < 8; ++j) {
      const int col = n0 + j * 16 + lm;
#pragma unroll
      for (int r = 0; r < 8; ++r) {
        const int row = rbase + r;
        if (OUTMODE == 0)
          ((float*)Cout)[(long)row * N + col] = acc[t][j][r] + bias[col];
        else if (OUTMODE == 1)
          ((_Float16*)Cout)[(long)row * N + col] = (_Float16)acc[t][j][r];
        else
          ((_Float16*)Cout)[(long)col * M + row] = (_Float16)acc[t][j][r];
      }
    }
  }
}

// ---------------------------------------------------------------------------
// Attention: one wave per (batch, head, 64-query block). L padded to 128.
// K tile [128][64] and transposed-V tile [64][128] are staged into LDS ONCE
// per block via global_load_async_to_lds_b128 (ASYNCcnt), then reused by all
// 4 query tiles. Q [B*S][1280] f16, Kb [B*128][1280] f16,
// Vt [1280][B*128] f16, O [B*S][1280] f16.
// ---------------------------------------------------------------------------
__global__ void __launch_bounds__(32)
attn_wmma(const _Float16* __restrict__ Q, const _Float16* __restrict__ Kb,
          const _Float16* __restrict__ Vt, _Float16* __restrict__ O) {
  __shared__ __align__(32) _Float16 Klds[LP * DH];   // [key l][d]  16KB
  __shared__ __align__(32) _Float16 Vlds[DH * LP];   // [d][key l]  16KB
  __shared__ __align__(32) _Float16 Plds[16 * LP];   //              4KB

  const int lane  = threadIdx.x;
  const int lm    = lane & 15;
  const int half  = lane >> 4;
  const int kbase = half << 4;
  const int koff  = half << 4;
  const int b     = blockIdx.y / NHEADS;
  const int h     = blockIdx.y % NHEADS;

  // ---- async-stage K and Vt tiles (1024 x 16B chunks each) into LDS ----
  const _Float16* kg = Kb + (long)(b * LP) * HDIM + h * DH;
  const _Float16* vg = Vt + (long)(h * DH) * (NB * LP) + b * LP;
  const unsigned kds0 = (unsigned)(unsigned long long)(&Klds[0]);
  const unsigned vds0 = (unsigned)(unsigned long long)(&Vlds[0]);
  for (int i = 0; i < 32; ++i) {
    const int c = i * 32 + lane;            // 0..1023
    {
      const int row = c >> 3, sub = c & 7;  // 8 chunks per 64-half K row
      const _Float16* src = kg + (long)row * HDIM + sub * 8;
      const unsigned dst = kds0 + c * 16;
      asm volatile("global_load_async_to_lds_b128 %0, %1, off"
                   :: "v"(dst), "v"(src) : "memory");
    }
    {
      const int row = c >> 4, sub = c & 15; // 16 chunks per 128-half Vt row
      const _Float16* src = vg + (long)row * (NB * LP) + sub * 8;
      const unsigned dst = vds0 + c * 16;
      asm volatile("global_load_async_to_lds_b128 %0, %1, off"
                   :: "v"(dst), "v"(src) : "memory");
    }
  }
  asm volatile("s_wait_asynccnt 0x0" ::: "memory");
  __syncthreads();

  const float SCALE = 0.125f;  // 1/sqrt(64)

  for (int qt = 0; qt < 4; ++qt) {
    const int qs = blockIdx.x * 64 + qt * 16;

    // ---- scores = Q @ K^T : 2 K-steps over dh, 8 N-tiles over L ----
    v8f sc[8];
#pragma unroll
    for (int j = 0; j < 8; ++j)
#pragma unroll
      for (int i = 0; i < 8; ++i) sc[j][i] = 0.0f;

#pragma unroll
    for (int ks = 0; ks < 2; ++ks) {
      const _Float16* qrow =
          Q + (long)(b * S_LEN + qs + lm) * HDIM + h * DH + ks * 32;
      const v16h a = make_a(*(const v8h*)(qrow + kbase),
                            *(const v8h*)(qrow + kbase + 8));
#pragma unroll
      for (int j = 0; j < 8; ++j) {
        const v16h kf =
            *(const v16h*)&Klds[(j * 16 + lm) * DH + ks * 32 + koff];
        sc[j] = __builtin_amdgcn_wmma_f32_16x16x32_f16(
            false, a, false, kf, (short)0, sc[j], false, false);
      }
    }

    // ---- softmax over L (mask l >= 109); row reductions stay inside the
    //      16-lane halves (shfl_xor masks 1/2/4/8 never cross the boundary) --
    float rinv[8];
#pragma unroll
    for (int r = 0; r < 8; ++r) {
      float mx = -1e30f;
#pragma unroll
      for (int j = 0; j < 8; ++j) {
        const float s = sc[j][r] * SCALE;
        sc[j][r] = s;
        mx = (j * 16 + lm < LVALID) ? fmaxf(mx, s) : mx;
      }
      mx = fmaxf(mx, __shfl_xor(mx, 8, 32));
      mx = fmaxf(mx, __shfl_xor(mx, 4, 32));
      mx = fmaxf(mx, __shfl_xor(mx, 2, 32));
      mx = fmaxf(mx, __shfl_xor(mx, 1, 32));

      float sum = 0.0f;
      const int m = r + (half << 3);
#pragma unroll
      for (int j = 0; j < 8; ++j) {
        const int n = j * 16 + lm;
        const float p = (n < LVALID) ? __expf(sc[j][r] - mx) : 0.0f;
        sum += p;
        Plds[m * LP + n] = (_Float16)p;
      }
      sum += __shfl_xor(sum, 8, 32);
      sum += __shfl_xor(sum, 4, 32);
      sum += __shfl_xor(sum, 2, 32);
      sum += __shfl_xor(sum, 1, 32);
      rinv[r] = 1.0f / sum;
    }
    __syncthreads();

    // ---- out = P @ V : 4 K-steps over L, 4 N-tiles over dh ----
    v8f oacc[4];
#pragma unroll
    for (int j = 0; j < 4; ++j)
#pragma unroll
      for (int i = 0; i < 8; ++i) oacc[j][i] = 0.0f;

#pragma unroll
    for (int kk = 0; kk < 4; ++kk) {
      const v16h a =
          make_a(*(const v8h*)&Plds[lm * LP + kk * 32 + kbase],
                 *(const v8h*)&Plds[lm * LP + kk * 32 + kbase + 8]);
#pragma unroll
      for (int jn = 0; jn < 4; ++jn) {
        const v16h vf =
            *(const v16h*)&Vlds[(jn * 16 + lm) * LP + kk * 32 + koff];
        oacc[jn] = __builtin_amdgcn_wmma_f32_16x16x32_f16(
            false, a, false, vf, (short)0, oacc[jn], false, false);
      }
    }

#pragma unroll
    for (int jn = 0; jn < 4; ++jn) {
#pragma unroll
      for (int r = 0; r < 8; ++r) {
        const int row = qs + r + (half << 3);
        const int col = h * DH + jn * 16 + lm;
        O[(long)(b * S_LEN + row) * HDIM + col] =
            (_Float16)(oacc[jn][r] * rinv[r]);
      }
    }
    __syncthreads();  // WAR: Plds reused by the next query tile
  }
}

// ---------------------------------------------------------------------------
extern "C" void kernel_launch(void* const* d_in, const int* in_sizes, int n_in,
                              void* d_out, int out_size, void* d_ws, size_t ws_size,
                              hipStream_t stream) {
  (void)in_sizes; (void)n_in; (void)out_size; (void)ws_size;
  const float* hs    = (const float*)d_in[0];
  const float* ehs   = (const float*)d_in[1];
  const float* idemb = (const float*)d_in[2];
  const float* Wq    = (const float*)d_in[3];
  const float* Wk    = (const float*)d_in[4];
  const float* Wv    = (const float*)d_in[5];
  const float* Widk  = (const float*)d_in[6];
  const float* Widv  = (const float*)d_in[7];
  const float* Wout  = (const float*)d_in[8];
  const float* bout  = (const float*)d_in[9];
  float* out = (float*)d_out;

  char* ws = (char*)d_ws;
  size_t off = 0;
  auto alloc = [&](size_t bytes) {
    char* p = ws + off;
    off += (bytes + 255) & ~(size_t)255;
    return p;
  };
  const long MQ = (long)NB * S_LEN;                         // 16384
  _Float16* hsF  = (_Float16*)alloc(MQ * HDIM * 2);         // reused as attn out
  _Float16* Qbuf = (_Float16*)alloc(MQ * HDIM * 2);
  _Float16* WqT  = (_Float16*)alloc((long)HDIM * HDIM * 2);
  _Float16* WoT  = (_Float16*)alloc((long)HDIM * HDIM * 2);
  _Float16* kvA  = (_Float16*)alloc((long)NB * LP * 4096 * 2);
  _Float16* WkT  = (_Float16*)alloc((long)HDIM * 4096 * 2);
  _Float16* WvT  = (_Float16*)alloc((long)HDIM * 4096 * 2);
  _Float16* Kbuf = (_Float16*)alloc((long)NB * LP * HDIM * 2);
  _Float16* Vt   = (_Float16*)alloc((long)NB * LP * HDIM * 2);

  const int TPB = 256;
  // ---- packing ----
  f32_to_f16_kernel<<<(unsigned)((MQ * HDIM + TPB - 1) / TPB), TPB, 0, stream>>>(
      hs, hsF, MQ * HDIM);
  transT_f16_kernel<<<(unsigned)(((long)HDIM * HDIM + TPB - 1) / TPB), TPB, 0, stream>>>(
      Wq, WqT, HDIM, HDIM);
  transT_f16_kernel<<<(unsigned)(((long)HDIM * HDIM + TPB - 1) / TPB), TPB, 0, stream>>>(
      Wout, WoT, HDIM, HDIM);
  kva_pack_kernel<<<(unsigned)(((long)NB * LP * 4096 + TPB - 1) / TPB), TPB, 0, stream>>>(
      ehs, idemb, kvA);
  catT_f16_kernel<<<(unsigned)(((long)HDIM * 4096 + TPB - 1) / TPB), TPB, 0, stream>>>(
      Wk, Widk, WkT);
  catT_f16_kernel<<<(unsigned)(((long)HDIM * 4096 + TPB - 1) / TPB), TPB, 0, stream>>>(
      Wv, Widv, WvT);

  // ---- Q = hs @ Wq (f16 out) ----
  gemm_wmma<1><<<dim3(512, 10), 32, 0, stream>>>(hsF, WqT, Qbuf, nullptr,
                                                 (int)MQ, HDIM, HDIM);
  // ---- K = kvA @ [Wk;Wid_k] (f16, row-major [B*128][1280]) ----
  gemm_wmma<1><<<dim3(16, 10), 32, 0, stream>>>(kvA, WkT, Kbuf, nullptr,
                                                NB * LP, HDIM, 4096);
  // ---- V = kvA @ [Wv;Wid_v] (f16, transposed -> Vt [1280][B*128]) ----
  gemm_wmma<2><<<dim3(16, 10), 32, 0, stream>>>(kvA, WvT, Vt, nullptr,
                                                NB * LP, HDIM, 4096);
  // ---- attention (writes f16 into hsF) ----
  attn_wmma<<<dim3(S_LEN / 64, NB * NHEADS), 32, 0, stream>>>(Qbuf, Kbuf, Vt, hsF);
  // ---- out = attn @ Wout + bout (f32) ----
  gemm_wmma<0><<<dim3(512, 10), 32, 0, stream>>>(hsF, WoT, out, bout,
                                                 (int)MQ, HDIM, HDIM);
}
